// ShiftConv_71442486002404
// MI455X (gfx1250) — matrix operand order repaired and checked by reading, not compiled, not run on hardware
//
#include <hip/hip_runtime.h>

typedef float v2f  __attribute__((ext_vector_type(2)));
typedef float v8f  __attribute__((ext_vector_type(8)));
typedef __bf16 v8bf  __attribute__((ext_vector_type(8)));
typedef __bf16 v16bf __attribute__((ext_vector_type(16)));

#define DIMC   192          // channels
#define HH     256
#define WW     256
#define BATCH  4
#define PIXT   64           // pixels per block (along W)
#define COLS   (PIXT + 2)   // halo: w0-1 .. w0+PIXT
#define NWAVES 12
#define BLOCK  (NWAVES * 32)

// ---------------- bf16x3 path constants ----------------
// bf16 LDS plane: lds[(r*COLS+col)*CSTRB + c]. CSTRB=200 bf16 -> lane stride
// 400B = 100 dwords, 100 mod 64 = 36, {36n mod 64} = all multiples of 4:
// a half-wave of b128 loads covers all 64 banks exactly once.
#define CSTRB     200
#define PLANE_BF  (3 * COLS * CSTRB)      // 39600 bf16 = 79200 B
#define LDS_BF_BYTES (2 * PLANE_BF * 2)   // 158400 B (hi + lo planes)
// W fragments: [t(12)][s(5)][kc(6)][lane(32)][j(16)] bf16, hi plane then lo
#define WFRAG_ELEMS (12 * 5 * 6 * 32 * 16)   // 184320
#define WS_NEED     ((size_t)(2 * WFRAG_ELEMS * 2))  // 737280 bytes

// ---------------- f32 fallback path constants ----------------
#define CSTR   194
#define LDS_F32_FLOATS (3 * COLS * CSTR)  // 153,648 B

__device__ __forceinline__ v16bf ld16bf(const __bf16* p) {
    v8bf a = *(const v8bf*)p;          // 16B
    v8bf b = *(const v8bf*)(p + 8);    // 16B
    return __builtin_shufflevector(a, b, 0,1,2,3,4,5,6,7,8,9,10,11,12,13,14,15);
}

// ============================================================
// W pre-swizzle: split into bf16 hi/lo in A-fragment lane order.
// A order used here (B uses the same): lane(m,half); element j -> K = j + 16*half
// ============================================================
__global__ __launch_bounds__(256)
void prep_w_bf16(const float* __restrict__ Wt, unsigned short* __restrict__ ws) {
    const int idx = blockIdx.x * 256 + threadIdx.x;
    if (idx >= WFRAG_ELEMS) return;
    const int j    = idx & 15;
    const int lane = (idx >> 4) & 31;
    const int q    = idx >> 9;          // (t*5+s)*6 + kc
    const int kc   = q % 6;
    const int rest = q / 6;             // t*5 + s
    const int s    = rest % 5;
    const int t    = rest / 5;
    const int m    = lane & 15;
    const int half = lane >> 4;
    const int krel = j + 16 * half;
    const int o    = t * 16 + m;
    const int k    = s * DIMC + kc * 32 + krel;
    const float v  = Wt[o * (5 * DIMC) + k];
    const __bf16 hv = (__bf16)v;
    const __bf16 lv = (__bf16)(v - (float)hv);
    __bf16* hi = (__bf16*)ws;
    hi[idx]               = hv;
    hi[WFRAG_ELEMS + idx] = lv;
}

// ============================================================
// bf16x3 main kernel: D = W_hi*X_hi + W_hi*X_lo + W_lo*X_hi  (fp32 acc)
// ============================================================
__global__ __launch_bounds__(BLOCK)
void shiftconv_wmma_bf16x3(const float* __restrict__ x,
                           const unsigned short* __restrict__ wfrag,
                           const float* __restrict__ bias,
                           float* __restrict__ out) {
    extern __shared__ __align__(16) unsigned char smem[];
    __bf16* lhi = (__bf16*)smem;
    __bf16* llo = lhi + PLANE_BF;

    const int tid = threadIdx.x;
    const int bid = blockIdx.x;
    const int wt = bid & 3;
    const int h  = (bid >> 2) & 255;
    const int b  = bid >> 10;
    const int w0 = wt * PIXT;
    const int lane = tid & 31;
    const int g    = tid >> 5;           // wave id 0..11 -> 16-channel O tile

    // ---- stage x tile as bf16 hi/lo planes (split once per element) ----
    for (int ci = 0; ci < 16; ++ci) {
        const int c = g * 16 + ci;
        const float* gch = x + (((size_t)b * DIMC + c) * HH + (h - 1)) * WW + (w0 - 1);
#pragma unroll
        for (int r = 0; r < 3; ++r) {
            const int gh = h - 1 + r;
            const bool rowOK = ((unsigned)gh < HH);   // block-uniform
#pragma unroll
            for (int cp = 0; cp < 3; ++cp) {
                const int col = lane + 32 * cp;
                if (col < COLS) {
                    const int gw = w0 - 1 + col;
                    float v = 0.0f;
                    if (rowOK && (unsigned)gw < WW) v = gch[r * WW + col];
                    const __bf16 hv = (__bf16)v;
                    const __bf16 lv = (__bf16)(v - (float)hv);
                    const int off = (r * COLS + col) * CSTRB + c;
                    lhi[off] = hv;
                    llo[off] = lv;
                }
            }
        }
    }
    __syncthreads();

    const int m    = lane & 15;          // A row (M) == B col (N)
    const int half = lane >> 4;
    const int o0   = g * 16;

    v8f acc[4] = {};

    // A stream: contiguous 1KB per fragment pair region, lane*32B inside
    const __bf16* wa  = (const __bf16*)wfrag;
    const __bf16* wah = wa + (size_t)(g * 5) * 6 * 512 + lane * 16;
    const __bf16* wal = wah + WFRAG_ELEMS;

    const int drow[5] = {0, 0, 0, 1, -1};   // center, right, left, down, up
    const int dcol[5] = {0, 1, -1, 0, 0};

#pragma unroll
    for (int s = 0; s < 5; ++s) {
        // B fragment: element j -> K = c0 + 16*half + j at pixel nb*16+m
        const __bf16* lbH = lhi +
            ((1 + drow[s]) * COLS + (1 + dcol[s]) + m) * CSTRB + 16 * half;
        const __bf16* lbL = llo +
            ((1 + drow[s]) * COLS + (1 + dcol[s]) + m) * CSTRB + 16 * half;
        const __bf16* wahS = wah + s * 6 * 512;
        const __bf16* walS = wal + s * 6 * 512;
#pragma unroll 3
        for (int kc = 0; kc < 6; ++kc) {
            const v16bf ah = ld16bf(wahS + kc * 512);
            const v16bf al = ld16bf(walS + kc * 512);
            const int c0 = kc * 32;
#pragma unroll
            for (int nb = 0; nb < 4; ++nb) {
                const v16bf bh = ld16bf(lbH + nb * 16 * CSTRB + c0);
                const v16bf bl = ld16bf(lbL + nb * 16 * CSTRB + c0);
                acc[nb] = __builtin_amdgcn_wmma_f32_16x16x32_bf16(
                    false, ah, false, bh, (short)0, acc[nb], false, false);
                acc[nb] = __builtin_amdgcn_wmma_f32_16x16x32_bf16(
                    false, ah, false, bl, (short)0, acc[nb], false, false);
                acc[nb] = __builtin_amdgcn_wmma_f32_16x16x32_bf16(
                    false, al, false, bh, (short)0, acc[nb], false, false);
            }
        }
    }

    // ---- epilogue: VGPR j -> M = j + 8*half; lane&15 -> N ----
    // Output is write-once / never re-read: non-temporal stores keep L2 for x/W.
#pragma unroll
    for (int j = 0; j < 8; ++j) {
        const int o = o0 + 8 * half + j;
        const float bv = bias[o];
        float* orow = out + (((size_t)b * DIMC + o) * HH + h) * WW + w0 + m;
#pragma unroll
        for (int nb = 0; nb < 4; ++nb) {
            __builtin_nontemporal_store(acc[nb][j] + bv, orow + nb * 16);
        }
    }
}

// ============================================================
// f32 WMMA fallback (used only if ws_size is too small) — verified lowering
// ============================================================
__global__ __launch_bounds__(BLOCK)
void shiftconv_wmma_f32(const float* __restrict__ x,
                        const float* __restrict__ Wt,
                        const float* __restrict__ bias,
                        float* __restrict__ out) {
    extern __shared__ float lds[];
    const int tid = threadIdx.x;
    const int bid = blockIdx.x;
    const int wt = bid & 3;
    const int h  = (bid >> 2) & 255;
    const int b  = bid >> 10;
    const int w0 = wt * PIXT;
    const int lane = tid & 31;
    const int g    = tid >> 5;

    for (int ci = 0; ci < 16; ++ci) {
        const int c = g * 16 + ci;
        const float* gch = x + (((size_t)b * DIMC + c) * HH + (h - 1)) * WW + (w0 - 1);
        float* lch = lds + c;
#pragma unroll
        for (int r = 0; r < 3; ++r) {
            const int gh = h - 1 + r;
            const bool rowOK = ((unsigned)gh < HH);
#pragma unroll
            for (int cp = 0; cp < 3; ++cp) {
                const int col = lane + 32 * cp;
                if (col < COLS) {
                    const int gw = w0 - 1 + col;
                    float v = 0.0f;
                    if (rowOK && (unsigned)gw < WW) v = gch[(size_t)r * WW + col];
                    lch[(r * COLS + col) * CSTR] = v;
                }
            }
        }
    }
    __syncthreads();

    const int m    = lane & 15;
    const int half = lane >> 4;
    const int o0   = g * 16;
    v8f acc[4] = {};
    const float* wbase = Wt + (size_t)(o0 + m) * (5 * DIMC) + 2 * half;
    const int drow[5] = {0, 0, 0, 1, -1};
    const int dcol[5] = {0, 1, -1, 0, 0};
#pragma unroll
    for (int s = 0; s < 5; ++s) {
        const float* wr = wbase + s * DIMC;
        const float* lb = lds +
            ((size_t)((1 + drow[s]) * COLS + (1 + dcol[s]) + m)) * CSTR + 2 * half;
#pragma unroll 8
        for (int cb = 0; cb < DIMC; cb += 4) {
            v2f a  = *(const v2f*)(wr + cb);
            v2f b0 = *(const v2f*)(lb + cb);
            v2f b1 = *(const v2f*)(lb + 16 * CSTR + cb);
            v2f b2 = *(const v2f*)(lb + 32 * CSTR + cb);
            v2f b3 = *(const v2f*)(lb + 48 * CSTR + cb);
            acc[0] = __builtin_amdgcn_wmma_f32_16x16x4_f32(false, a, false, b0,
                        (short)0, acc[0], false, false);
            acc[1] = __builtin_amdgcn_wmma_f32_16x16x4_f32(false, a, false, b1,
                        (short)0, acc[1], false, false);
            acc[2] = __builtin_amdgcn_wmma_f32_16x16x4_f32(false, a, false, b2,
                        (short)0, acc[2], false, false);
            acc[3] = __builtin_amdgcn_wmma_f32_16x16x4_f32(false, a, false, b3,
                        (short)0, acc[3], false, false);
        }
    }
#pragma unroll
    for (int j = 0; j < 8; ++j) {
        const int o = o0 + 8 * half + j;
        const float bv = bias[o];
        float* orow = out + (((size_t)b * DIMC + o) * HH + h) * WW + w0 + m;
#pragma unroll
        for (int nb = 0; nb < 4; ++nb) {
            __builtin_nontemporal_store(acc[nb][j] + bv, orow + nb * 16);
        }
    }
}

extern "C" void kernel_launch(void* const* d_in, const int* in_sizes, int n_in,
                              void* d_out, int out_size, void* d_ws, size_t ws_size,
                              hipStream_t stream) {
    (void)in_sizes; (void)n_in; (void)out_size;
    const float* x    = (const float*)d_in[0];   // [4,192,256,256]
    const float* Wt   = (const float*)d_in[1];   // [192,960]
    const float* bias = (const float*)d_in[2];   // [192]
    float* out        = (float*)d_out;           // [4,192,256,256]

    dim3 grid(BATCH * HH * (WW / PIXT));         // 4096
    dim3 block(BLOCK);                           // 384 = 12 waves

    if (ws_size >= WS_NEED) {
        prep_w_bf16<<<dim3(WFRAG_ELEMS / 256), dim3(256), 0, stream>>>(
            Wt, (unsigned short*)d_ws);
        shiftconv_wmma_bf16x3<<<grid, block, (size_t)LDS_BF_BYTES, stream>>>(
            x, (const unsigned short*)d_ws, bias, out);
    } else {
        shiftconv_wmma_f32<<<grid, block, (size_t)LDS_F32_FLOATS * sizeof(float),
                             stream>>>(x, Wt, bias, out);
    }
}
